// FBOTAM_64226940944777
// MI455X (gfx1250) — matrix operand order repaired and checked by reading, not compiled, not run on hardware
//
#include <hip/hip_runtime.h>

// ---------------------------------------------------------------------------
// MI455X (gfx1250, wave32) temporal-adaptive module.
//   K0: global branch -> per-(n,c) 3-tap kernels (VALU, tiny)
//   K1: repack l_w1 / l_w2 to bf16 GEMM layouts + fold conv BN scale/bias
//   K2: conv1d as bf16 WMMA GEMM (M=512, K=6144 folded taps, N=400) + BN + ReLU
//   K3: gate GEMM (M=2048, K=512, N=400) + sigmoid * lt -> new_lt
//   K4: 3-tap depthwise temporal aggregation (memory bound)
// Round-2 changes: branchless clamped B staging (no exec divergence in the
// K-loop), async A-tile staging via GLOBAL_LOAD_ASYNC_TO_LDS_B128 (+ASYNCcnt
// wait) overlapped with B-tile f32->bf16 transpose, folded BN constants.
// ---------------------------------------------------------------------------

typedef __attribute__((ext_vector_type(16))) __bf16 v16bf;
typedef __attribute__((ext_vector_type(8)))  float  v8f;

#define EPSV 1e-5f

constexpr int Ndim = 32, Cdim = 2048, STdim = 16, LTdim = 400;
constexpr int HID = 64, KT = 3, CB = 512;
constexpr int KCONV = KT * Cdim;           // 6144 folded reduction for conv GEMM
constexpr int ROWB = 40;                   // LDS row stride in bf16 (80 B: 16B-aligned, bank-spread)

// ---- async copy: 32 B global -> LDS per lane (two b128 beats, shared addr pair;
//      ISA adds INST_OFFSET to both LDS and global addresses)
__device__ __forceinline__ void async_copy32(unsigned lds_addr, unsigned gbyte_off,
                                             unsigned long long base)
{
    asm volatile("global_load_async_to_lds_b128 %0, %1, %2\n\t"
                 "global_load_async_to_lds_b128 %0, %1, %2 offset:16"
                 :: "v"(lds_addr), "v"(gbyte_off), "s"(base) : "memory");
}
__device__ __forceinline__ void wait_async0()
{
    asm volatile("s_wait_asynccnt 0x0" ::: "memory");
}
__device__ __forceinline__ unsigned lds_off(const void* p)
{
    return (unsigned)(uintptr_t)p;        // flat ptr low 32 bits == LDS byte offset
}

// ---------------------------------------------------------------- K0: global branch
__global__ void __launch_bounds__(256) k0_global_branch(
    const float* __restrict__ st, const float* __restrict__ w1,
    const float* __restrict__ gamma, const float* __restrict__ beta,
    const float* __restrict__ mean, const float* __restrict__ var,
    const float* __restrict__ w2, float* __restrict__ kern)
{
    __shared__ float sW1[HID * STdim];
    __shared__ float sS[HID], sB[HID];
    __shared__ float sW2[KT * HID];
    for (int i = threadIdx.x; i < HID * STdim; i += 256) sW1[i] = w1[i];
    for (int i = threadIdx.x; i < HID; i += 256) {
        float inv = rsqrtf(var[i] + EPSV) * gamma[i];
        sS[i] = inv;
        sB[i] = beta[i] - mean[i] * inv;
    }
    for (int i = threadIdx.x; i < KT * HID; i += 256) sW2[i] = w2[i];
    __syncthreads();

    int nc = blockIdx.x * 256 + threadIdx.x;          // exactly N*C = 65536 threads
    float x[STdim];
    const float* sp = st + nc * STdim;
#pragma unroll
    for (int i = 0; i < STdim; ++i) x[i] = sp[i];
    float z0 = 0.f, z1 = 0.f, z2 = 0.f;
#pragma unroll 4
    for (int j = 0; j < HID; ++j) {
        float h = 0.f;
#pragma unroll
        for (int i = 0; i < STdim; ++i) h += x[i] * sW1[j * STdim + i];
        h = fmaxf(h * sS[j] + sB[j], 0.f);
        z0 += h * sW2[0 * HID + j];
        z1 += h * sW2[1 * HID + j];
        z2 += h * sW2[2 * HID + j];
    }
    float m = fmaxf(z0, fmaxf(z1, z2));
    float e0 = __expf(z0 - m), e1 = __expf(z1 - m), e2 = __expf(z2 - m);
    float r = 1.f / (e0 + e1 + e2);
    float* kp = kern + (size_t)nc * 3;
    kp[0] = e0 * r; kp[1] = e1 * r; kp[2] = e2 * r;
}

// ---------------------------------------------------------------- K1: weight repack
// w1b[o][tap*2048 + c] = bf16(l_w1[o][c][tap])  -> conv GEMM A, K=6144, tap-major
__global__ void __launch_bounds__(256) k1_pack_w1(const float* __restrict__ w1,
                                                  __bf16* __restrict__ w1b)
{
    int idx = blockIdx.x * 256 + threadIdx.x;         // < 512*6144
    int o   = idx / KCONV;
    int j   = idx - o * KCONV;
    int tap = j >> 11;                                // j / 2048
    int c   = j & 2047;
    w1b[idx] = (__bf16)w1[(o * Cdim + c) * 3 + tap];
}

// also folds conv-branch BN into per-channel scale/bias
__global__ void __launch_bounds__(256) k1_pack_w2(
    const float* __restrict__ w2, __bf16* __restrict__ w2b,
    const float* __restrict__ gamma, const float* __restrict__ beta,
    const float* __restrict__ mean,  const float* __restrict__ var,
    float* __restrict__ scale, float* __restrict__ bias)
{
    int idx = blockIdx.x * 256 + threadIdx.x;         // < 2048*512
    w2b[idx] = (__bf16)w2[idx];
    if (idx < CB) {
        float s = rsqrtf(var[idx] + EPSV) * gamma[idx];
        scale[idx] = s;
        bias[idx]  = beta[idx] - mean[idx] * s;
    }
}

// ------------------------------------------------------- WMMA fragment helpers
// LDS tiles are K-contiguous with row stride ROWB (80 B). Fragment loads follow the
// CDNA5 16-bit A/B VGPR layouts: two aligned 16 B LDS reads per fragment.
__device__ __forceinline__ v16bf load_frag_A(const __bf16* tile, int row, int half)
{
    const __bf16* rp = tile + row * ROWB;
    union { uint4 u[2]; v16bf v; } t;
    t.u[0] = *(const uint4*)(rp + half * 8);          // K = half*8 .. +7   -> V0-3
    t.u[1] = *(const uint4*)(rp + 16 + half * 8);     // K = 16+half*8 ..  -> V4-7
    return t.v;
}
__device__ __forceinline__ v16bf load_frag_B(const __bf16* tile, int row, int half)
{
    const __bf16* rp = tile + row * ROWB;             // K = half*8 .. +7, 16+half*8 .. +7
    union { uint4 u[2]; v16bf v; } t;
    t.u[0] = *(const uint4*)(rp + half * 8);
    t.u[1] = *(const uint4*)(rp + 16 + half * 8);
    return t.v;
}

// ---------------------------------------------------------------- K2: conv GEMM
// x[n,o,t] = relu(BN( sum_{tap,c} w1[o,c,tap] * lt[n,c,t+tap-1] ))  -> bf16 xb
__global__ void __launch_bounds__(256) k2_conv_gemm(
    const float* __restrict__ lt, const __bf16* __restrict__ w1b,
    const float* __restrict__ scale, const float* __restrict__ bias,
    __bf16* __restrict__ xb)
{
    __shared__ __bf16 At[128 * ROWB];   // 128(M) x 32(K) bf16, padded rows
    __shared__ __bf16 Bt[64 * ROWB];    // 64(N=t) x 32(K=c) bf16, padded rows

    const int n  = blockIdx.z;
    const int m0 = blockIdx.x * 128;    // output-channel block (4 blocks over 512)
    const int t0 = blockIdx.y * 64;     // time block (7 blocks over 400)
    const int tid = threadIdx.x;
    const int lane = tid & 31, wid = tid >> 5;
    const int waveM = wid & 3, waveT = wid >> 2;      // 4x2 wave grid -> 32x32 per wave
    const int half = lane >> 4, lm = lane & 15;

    const float* ltn = lt + (size_t)n * Cdim * LTdim;
    const int arow = tid >> 1, ah = tid & 1;          // A staging: 2 threads/row, 32 B each
    const unsigned a_lds = lds_off(At + arow * ROWB + ah * 16);
    const unsigned long long a_base = (unsigned long long)(uintptr_t)w1b;
    v8f acc[2][2] = {};

    constexpr int STEPS = KCONV / 32;                 // 192
    for (int step = 0; step < STEPS; ++step) {
        const int j0  = step * 32;
        const int tap = j0 >> 11;                     // uniform per step (2048 % 32 == 0)
        const int c0  = j0 & 2047;

        // ---- stage A tile asynchronously: w1b[m0..+127][j0..+31] -> LDS (ASYNCcnt)
        async_copy32(a_lds, (unsigned)(((m0 + arow) * KCONV + j0 + ah * 16) * 2), a_base);

        // ---- stage B tile (overlaps async A): lt -> bf16, transposed [t][c],
        //      branchless: unconditional clamped load + value select
#pragma unroll
        for (int e = 0; e < 8; ++e) {
            int idx = e * 256 + tid;
            int c = idx >> 6, tt = idx & 63;
            int tsrc = t0 + tt + tap - 1;
            int tcl  = min(max(tsrc, 0), LTdim - 1);
            float v = ltn[(c0 + c) * LTdim + tcl];
            v = ((unsigned)tsrc < (unsigned)LTdim) ? v : 0.f;
            Bt[tt * ROWB + c] = (__bf16)v;
        }
        wait_async0();
        __syncthreads();

        v16bf a[2], b[2];
#pragma unroll
        for (int ms = 0; ms < 2; ++ms) a[ms] = load_frag_A(At, waveM * 32 + ms * 16 + lm, half);
#pragma unroll
        for (int nt = 0; nt < 2; ++nt) b[nt] = load_frag_B(Bt, waveT * 32 + nt * 16 + lm, half);
#pragma unroll
        for (int ms = 0; ms < 2; ++ms)
#pragma unroll
            for (int nt = 0; nt < 2; ++nt)
                acc[ms][nt] = __builtin_amdgcn_wmma_f32_16x16x32_bf16(
                    false, a[ms], false, b[nt], (short)0, acc[ms][nt], false, false);
        __syncthreads();
    }

    // ---- epilogue: folded BN + ReLU, store bf16 x (C-layout: N=lane, M=vreg+8*half)
#pragma unroll
    for (int ms = 0; ms < 2; ++ms) {
#pragma unroll
        for (int nt = 0; nt < 2; ++nt) {
#pragma unroll
            for (int r = 0; r < 8; ++r) {
                int o = m0 + waveM * 32 + ms * 16 + r + half * 8;
                int t = t0 + waveT * 32 + nt * 16 + lm;
                if (t < LTdim) {
                    float v = fmaxf(acc[ms][nt][r] * scale[o] + bias[o], 0.f);
                    xb[((size_t)n * CB + o) * LTdim + t] = (__bf16)v;
                }
            }
        }
    }
}

// ---------------------------------------------------------------- K3: gate GEMM
// new_lt[n,c,t] = lt[n,c,t] * sigmoid( sum_o l_w2[c,o] * x[n,o,t] )
__global__ void __launch_bounds__(256) k3_gate_gemm(
    const float* __restrict__ lt, const __bf16* __restrict__ w2b,
    const __bf16* __restrict__ xb, float* __restrict__ nl)
{
    __shared__ __bf16 At[128 * ROWB];
    __shared__ __bf16 Bt[64 * ROWB];

    const int n  = blockIdx.z;
    const int m0 = blockIdx.x * 128;    // c block (16 blocks over 2048)
    const int t0 = blockIdx.y * 64;
    const int tid = threadIdx.x;
    const int lane = tid & 31, wid = tid >> 5;
    const int waveM = wid & 3, waveT = wid >> 2;
    const int half = lane >> 4, lm = lane & 15;

    const unsigned short* xs = (const unsigned short*)(xb + (size_t)n * CB * LTdim);
    unsigned short* Bs = (unsigned short*)Bt;
    const int arow = tid >> 1, ah = tid & 1;
    const unsigned a_lds = lds_off(At + arow * ROWB + ah * 16);
    const unsigned long long a_base = (unsigned long long)(uintptr_t)w2b;
    v8f acc[2][2] = {};

    constexpr int STEPS = CB / 32;                    // 16
    for (int step = 0; step < STEPS; ++step) {
        const int o0 = step * 32;
        // ---- stage A tile asynchronously: w2b[m0..+127][o0..+31] -> LDS
        async_copy32(a_lds, (unsigned)(((m0 + arow) * CB + o0 + ah * 16) * 2), a_base);

        // ---- stage B tile: x[n, o0..+31, t0..+63] bf16, transposed [t][o], branchless
#pragma unroll
        for (int e = 0; e < 8; ++e) {
            int idx = e * 256 + tid;
            int oo = idx >> 6, tt = idx & 63;
            int t = t0 + tt;
            int tcl = min(t, LTdim - 1);
            unsigned short uv = xs[(o0 + oo) * LTdim + tcl];
            uv = (t < LTdim) ? uv : (unsigned short)0;
            Bs[tt * ROWB + oo] = uv;
        }
        wait_async0();
        __syncthreads();

        v16bf a[2], b[2];
#pragma unroll
        for (int ms = 0; ms < 2; ++ms) a[ms] = load_frag_A(At, waveM * 32 + ms * 16 + lm, half);
#pragma unroll
        for (int nt = 0; nt < 2; ++nt) b[nt] = load_frag_B(Bt, waveT * 32 + nt * 16 + lm, half);
#pragma unroll
        for (int ms = 0; ms < 2; ++ms)
#pragma unroll
            for (int nt = 0; nt < 2; ++nt)
                acc[ms][nt] = __builtin_amdgcn_wmma_f32_16x16x32_bf16(
                    false, a[ms], false, b[nt], (short)0, acc[ms][nt], false, false);
        __syncthreads();
    }

    // ---- epilogue: sigmoid gate * lt -> new_lt (f32)
#pragma unroll
    for (int ms = 0; ms < 2; ++ms) {
#pragma unroll
        for (int nt = 0; nt < 2; ++nt) {
#pragma unroll
            for (int r = 0; r < 8; ++r) {
                int c = m0 + waveM * 32 + ms * 16 + r + half * 8;
                int t = t0 + waveT * 32 + nt * 16 + lm;
                if (t < LTdim) {
                    float g = 1.f / (1.f + __expf(-acc[ms][nt][r]));
                    size_t off = ((size_t)n * Cdim + c) * LTdim + t;
                    nl[off] = lt[off] * g;
                }
            }
        }
    }
}

// ---------------------------------------------------------------- K4: 3-tap aggregation
__global__ void __launch_bounds__(256) k4_aggregate(
    const float* __restrict__ nl, const float* __restrict__ kern,
    float* __restrict__ out)
{
    int idx = blockIdx.x * 256 + threadIdx.x;         // exactly N*C*LT threads
    int t  = idx % LTdim;
    int nc = idx / LTdim;
    const float* kp = kern + (size_t)nc * 3;
    const float* p  = nl + (size_t)nc * LTdim;
    float left  = (t > 0)          ? p[t - 1] : 0.f;
    float mid   = p[t];
    float right = (t < LTdim - 1)  ? p[t + 1] : 0.f;
    out[idx] = kp[0] * left + kp[1] * mid + kp[2] * right;
}

// ---------------------------------------------------------------- launch
extern "C" void kernel_launch(void* const* d_in, const int* in_sizes, int n_in,
                              void* d_out, int out_size, void* d_ws, size_t ws_size,
                              hipStream_t stream)
{
    const float* st_feat = (const float*)d_in[0];
    const float* lt_feat = (const float*)d_in[1];
    const float* g_w1    = (const float*)d_in[2];
    const float* g_gamma = (const float*)d_in[3];
    const float* g_beta  = (const float*)d_in[4];
    const float* g_mean  = (const float*)d_in[5];
    const float* g_var   = (const float*)d_in[6];
    const float* g_w2    = (const float*)d_in[7];
    const float* l_w1    = (const float*)d_in[8];
    const float* l_gamma = (const float*)d_in[9];
    const float* l_beta  = (const float*)d_in[10];
    const float* l_mean  = (const float*)d_in[11];
    const float* l_var   = (const float*)d_in[12];
    const float* l_w2    = (const float*)d_in[13];
    float* out = (float*)d_out;

    // workspace partition (~127 MB)
    char* w = (char*)d_ws;
    float*  kern   = (float*)w;  w += (size_t)Ndim * Cdim * 3 * 4;           // 0.79 MB
    float*  lscale = (float*)w;  w += CB * 4;                                // 2 KB
    float*  lbias  = (float*)w;  w += CB * 4;                                // 2 KB
    __bf16* w1b    = (__bf16*)w; w += (size_t)CB * KCONV * 2;                // 6.3 MB
    __bf16* w2b    = (__bf16*)w; w += (size_t)Cdim * CB * 2;                 // 2.1 MB
    __bf16* xbuf   = (__bf16*)w; w += (size_t)Ndim * CB * LTdim * 2;         // 13.1 MB (L2-resident)
    float*  nl     = (float*)w;                                              // 104.9 MB

    k0_global_branch<<<Ndim * Cdim / 256, 256, 0, stream>>>(
        st_feat, g_w1, g_gamma, g_beta, g_mean, g_var, g_w2, kern);
    k1_pack_w1<<<CB * KCONV / 256, 256, 0, stream>>>(l_w1, w1b);
    k1_pack_w2<<<Cdim * CB / 256, 256, 0, stream>>>(
        l_w2, w2b, l_gamma, l_beta, l_mean, l_var, lscale, lbias);

    dim3 g2(CB / 128, (LTdim + 63) / 64, Ndim);       // 4 x 7 x 32
    k2_conv_gemm<<<g2, 256, 0, stream>>>(lt_feat, w1b, lscale, lbias, xbuf);

    dim3 g3(Cdim / 128, (LTdim + 63) / 64, Ndim);     // 16 x 7 x 32
    k3_gate_gemm<<<g3, 256, 0, stream>>>(lt_feat, w2b, xbuf, nl);

    k4_aggregate<<<(size_t)Ndim * Cdim * LTdim / 256, 256, 0, stream>>>(nl, kern, out);
}
